// SoftIGNN_30064771072228
// MI455X (gfx1250) — matrix operand order repaired and checked by reading, not compiled, not run on hardware
//
#include <hip/hip_runtime.h>
#include <hip/hip_bf16.h>

#define NNODES 100000
#define NEDGES 3200000
#define IN_C   256
#define OUT_C  128
#define KAPPA  0.95f

typedef __attribute__((ext_vector_type(16))) __bf16 v16bf;
typedef __attribute__((ext_vector_type(8)))  float  v8f;

__device__ __forceinline__ __bf16 to_bf16(float f) { return static_cast<__bf16>(f); }

// ---------------- weight conversion ----------------
__global__ void cvt_bf16_kernel(const float* __restrict__ in, __bf16* __restrict__ out, int n) {
    int i = blockIdx.x * blockDim.x + threadIdx.x;
    if (i < n) out[i] = to_bf16(in[i]);
}

// ---------------- proj_norm_inf (one block of 128 threads per row) ----------------
__global__ void proj_kernel(const float* __restrict__ Wc, __bf16* __restrict__ Wp) {
    __shared__ float u[128];
    __shared__ float s[128];
    __shared__ int cnt;
    const int row = blockIdx.x;
    const int tid = threadIdx.x;

    float w  = Wc[row * 128 + tid];
    float av = fabsf(w);
    u[tid] = av;
    if (tid == 0) cnt = 0;
    __syncthreads();

    // bitonic sort, descending
    for (int k = 2; k <= 128; k <<= 1) {
        for (int j = k >> 1; j > 0; j >>= 1) {
            int ixj = tid ^ j;
            if (ixj > tid) {
                float x = u[tid], y = u[ixj];
                bool up = ((tid & k) == 0);
                bool sw = up ? (x < y) : (x > y);   // descending overall
                if (sw) { u[tid] = y; u[ixj] = x; }
            }
            __syncthreads();
        }
    }

    // inclusive scan (Hillis-Steele)
    s[tid] = u[tid];
    __syncthreads();
    for (int off = 1; off < 128; off <<= 1) {
        float add = (tid >= off) ? s[tid - off] : 0.0f;
        __syncthreads();
        s[tid] += add;
        __syncthreads();
    }

    float rowsum = s[127];
    float css    = s[tid] - KAPPA;
    bool  cond   = (u[tid] - css / (float)(tid + 1)) > 0.0f;
    if (cond) atomicAdd(&cnt, 1);
    __syncthreads();

    int   rho   = (cnt > 1) ? cnt : 1;
    float theta = (s[rho - 1] - KAPPA) / (float)rho;
    float sgn   = (w > 0.0f) ? 1.0f : ((w < 0.0f) ? -1.0f : 0.0f);
    float proj  = fmaxf(av - theta, 0.0f) * sgn;
    float res   = (rowsum > KAPPA) ? proj : w;
    Wp[row * 128 + tid] = to_bf16(res);
}

// ---------------- degree / norm ----------------
__global__ void deg_init_kernel(float* __restrict__ deg) {
    int i = blockIdx.x * blockDim.x + threadIdx.x;
    if (i < NNODES) deg[i] = 1.0f;          // self loop
}
__global__ void deg_count_kernel(const int* __restrict__ dst, float* __restrict__ deg) {
    int e = blockIdx.x * blockDim.x + threadIdx.x;
    if (e < NEDGES) atomicAdd(&deg[dst[e]], 1.0f);
}
__global__ void deg_fin_kernel(float* __restrict__ deg) {
    int i = blockIdx.x * blockDim.x + threadIdx.x;
    if (i < NNODES) deg[i] = rsqrtf(deg[i]);   // deg >= 1 always
}

// ---------------- WMMA fragment loaders ----------------
// A-matrix (16x32 bf16 from fp32 source): lanes 0-15 -> row M=lane, K in {k0..k0+7, k0+16..k0+23};
// lanes 16-31 -> same rows, K in {k0+8..k0+15, k0+24..k0+31}  (ISA 7.12.2 16-bit A layout)
__device__ __forceinline__ v16bf load_a_f32(const float* __restrict__ M, int ld,
                                            int r0, int k0, int lane) {
    int r  = r0 + (lane & 15);
    int kb = k0 + ((lane >> 4) << 3);
    const float* p = M + (size_t)r * ld + kb;
    v16bf f;
#pragma unroll
    for (int i = 0; i < 8; ++i) {
        f[i]     = to_bf16(p[i]);
        f[i + 8] = to_bf16(p[i + 16]);
    }
    return f;
}

struct B32bytes { uint4 lo, hi; };

// B-matrix (32x16 bf16), B[k][n] = W[n][k]: lanes 0-15 -> column n=lane, K in {k0..k0+7, k0+16..k0+23};
// lanes 16-31 -> column n=lane-16, K in {k0+8..k0+15, k0+24..k0+31} (ISA B layout pattern). Source = padded LDS.
template<int K>
__device__ __forceinline__ v16bf load_b_lds(const __bf16* W, int n0, int k0, int lane) {
    int r  = n0 + (lane & 15);
    int kb = k0 + ((lane >> 4) << 3);
    const __bf16* p = W + r * (K + 8) + kb;   // 16B-aligned (row pitch 2K+16 bytes)
    B32bytes t;
    t.lo = *(const uint4*)(p);
    t.hi = *(const uint4*)(p + 16);
    return __builtin_bit_cast(v16bf, t);
}

// ---------------- fused GEMM: D = A[N x K] @ W[128 x K]^T (+ optional self-loop term) ---------
// One wave per 16-row strip; weights staged in padded LDS once per 256-thread block.
template<int K, bool FUSE>
__global__ void __launch_bounds__(256) gemm_kernel(const float* __restrict__ A,
                                                   const __bf16* __restrict__ W,
                                                   const float* __restrict__ h,
                                                   const float* __restrict__ dinv,
                                                   float* __restrict__ dstm) {
    __shared__ __bf16 Wlds[128 * (K + 8)];
    {
        // cooperative copy, 8 bf16 (16B) per chunk, honoring the +8 bf16 row pad
        const int cpr = K / 8;
        for (int c = threadIdx.x; c < 128 * cpr; c += 256) {
            int r = c / cpr, o = (c - r * cpr) * 8;
            *(uint4*)&Wlds[r * (K + 8) + o] = *(const uint4*)&W[r * K + o];
        }
    }
    __syncthreads();

    const int lane = threadIdx.x & 31;
    const int wave = blockIdx.x * 8 + (threadIdx.x >> 5);
    if (wave >= NNODES / 16) return;       // wave-uniform: EXEC stays all-ones
    const int r0 = wave * 16;

    v8f acc[8] = {};
    v16bf a = load_a_f32(A, K, r0, 0, lane);
#pragma unroll
    for (int k0 = 0; k0 < K; k0 += 32) {
        // pull all 8 n-tile B-frags from LDS up front
        v16bf b[8];
#pragma unroll
        for (int nt = 0; nt < 8; ++nt)
            b[nt] = load_b_lds<K>(Wlds, nt * 16, k0, lane);
        // double-buffer A from global
        v16bf an = a;
        if (k0 + 32 < K) an = load_a_f32(A, K, r0, k0 + 32, lane);
#pragma unroll
        for (int nt = 0; nt < 8; ++nt)
            acc[nt] = __builtin_amdgcn_wmma_f32_16x16x32_bf16(
                false, a, false, b[nt], (short)0, acc[nt], false, false);
        a = an;
    }

    // C/D layout: VGPR r -> M = r (lanes 0-15) or r+8 (lanes 16-31); N = lane&15
    const int n  = lane & 15;
    const int mb = (lane >> 4) << 3;
#pragma unroll
    for (int nt = 0; nt < 8; ++nt) {
#pragma unroll
        for (int r = 0; r < 8; ++r) {
            int row = r0 + mb + r;
            size_t idx = (size_t)row * 128 + nt * 16 + n;
            if constexpr (FUSE) {
                float d = dinv[row];
                dstm[idx] = acc[nt][r] + h[idx] * d * d;   // + self-loop h[v]*dinv[v]^2
            } else {
                dstm[idx] = acc[nt][r];
            }
        }
    }
}

// ---------------- edge scatter: one wave per edge ----------------
__global__ void scatter_kernel(const int* __restrict__ src, const int* __restrict__ dst,
                               const float* __restrict__ h, const float* __restrict__ dinv,
                               float* __restrict__ out) {
    const int gid  = blockIdx.x * blockDim.x + threadIdx.x;
    const int e    = gid >> 5;
    const int lane = threadIdx.x & 31;
    if (e >= NEDGES) return;
    const int s = src[e];
    const int d = dst[e];
    const float w = dinv[s] * dinv[d];
    float4 v = ((const float4*)(h + (size_t)s * 128))[lane];
    float* op = out + (size_t)d * 128 + lane * 4;
    atomicAdd(op + 0, v.x * w);
    atomicAdd(op + 1, v.y * w);
    atomicAdd(op + 2, v.z * w);
    atomicAdd(op + 3, v.w * w);
}

// ---------------- final relu ----------------
__global__ void relu_kernel(float* __restrict__ out, int n) {
    int i = blockIdx.x * blockDim.x + threadIdx.x;
    if (i < n) out[i] = fmaxf(out[i], 0.0f);
}

extern "C" void kernel_launch(void* const* d_in, const int* in_sizes, int n_in,
                              void* d_out, int out_size, void* d_ws, size_t ws_size,
                              hipStream_t stream) {
    const float* features = (const float*)d_in[0];   // [N, 256]
    const int*   eidx     = (const int*)d_in[1];     // [2, E]
    const float* emb      = (const float*)d_in[2];   // [N, 128]
    const float* Wc       = (const float*)d_in[3];   // [128, 128]
    const float* Wm       = (const float*)d_in[4];   // [128, 256]
    float* out = (float*)d_out;                      // [N, 128]
    const int* src = eidx;
    const int* dst = eidx + NEDGES;

    char* ws = (char*)d_ws;
    size_t off = 0;
    float* h = (float*)(ws + off);          off += (size_t)NNODES * 128 * sizeof(float);
    float* deg = (float*)(ws + off);        off += (size_t)NNODES * sizeof(float);
    __bf16* Wpb = (__bf16*)(ws + off);      off += (size_t)128 * 128 * sizeof(__bf16);
    __bf16* Wmb = (__bf16*)(ws + off);      off += (size_t)128 * IN_C * sizeof(__bf16);
    (void)off; (void)ws_size; (void)n_in; (void)in_sizes; (void)out_size;

    cvt_bf16_kernel<<<(128 * IN_C) / 256, 256, 0, stream>>>(Wm, Wmb, 128 * IN_C);
    proj_kernel<<<128, 128, 0, stream>>>(Wc, Wpb);
    deg_init_kernel<<<(NNODES + 255) / 256, 256, 0, stream>>>(deg);
    deg_count_kernel<<<(NEDGES + 255) / 256, 256, 0, stream>>>(dst, deg);
    deg_fin_kernel<<<(NNODES + 255) / 256, 256, 0, stream>>>(deg);

    const int nwaves = NNODES / 16;                   // 6250
    const int gblocks = (nwaves + 7) / 8;             // 8 waves / 256-thread block
    gemm_kernel<128, false><<<gblocks, 256, 0, stream>>>(emb, Wpb, nullptr, nullptr, h);
    gemm_kernel<IN_C, true><<<gblocks, 256, 0, stream>>>(features, Wmb, h, deg, out);

    scatter_kernel<<<NEDGES / 8, 256, 0, stream>>>(src, dst, h, deg, out);
    relu_kernel<<<(NNODES * 128) / 256, 256, 0, stream>>>(out, NNODES * 128);
}
